// GENELink_71597104824562
// MI455X (gfx1250) — matrix operand order repaired and checked by reading, not compiled, not run on hardware
//
#include <hip/hip_runtime.h>
#include <hip/hip_bf16.h>

// ---------------------------------------------------------------------------
// GENELink GAT layer for MI455X (gfx1250, wave32, WMMA).
//
// Roofline: adjacency streams dominate (~576 MB int32 -> ~25us @ 23.3 TB/s).
// Attention GEMMs ~19.3 GFLOP done as v_wmma_f32_16x16x32_f16 (f32 accum),
// with the attention operand (adj ? exp(leaky(s1+s2)) : 0) generated in
// registers while streaming adjacency exactly once; the softmax denominator
// is accumulated in the same pass and applied in the epilogue.
//
// Inner loops use constant-stride pointer advancement so all hot loads are
// immediate-offset global_load_b128 (A runs at +0/+16/+64/+80; B frags at
// nt*1024+{0,16}), minimizing 64-bit address VALU per iteration.
// ---------------------------------------------------------------------------

#define NGn 8192
#define NCn 4096
#define FGk 512
#define FCk 256
#define DD  64
#define ALPHA 0.2f

typedef _Float16 h16;
typedef __attribute__((ext_vector_type(16))) _Float16 v16h;
typedef __attribute__((ext_vector_type(8)))  float    v8f;
typedef __attribute__((ext_vector_type(4)))  float    v4f;
typedef __attribute__((ext_vector_type(4)))  int      v4i;

// A-fragment 16-bit layout (ISA 7.12.2): lane = (row r in 0..15) + 16*hi.
// VGPR v0..3 hold K = 8*hi + {0..7}; v4..7 hold K = 16 + 8*hi + {0..7}.
// => per lane, K is two contiguous runs of 8: [8*hi, +8) and [16+8*hi, +8).
__device__ __forceinline__ int koffA(int hi, int idx) {
  return (idx < 8) ? (8 * hi + idx) : (16 + 8 * hi + (idx - 8));
}
// Contiguous k order (unused; B frags here use koffA): lanes 0-15 hold K=0..15,
// lanes 16-31 hold K=16..31, column N = lane & 15.
__device__ __forceinline__ int koffB(int hi, int idx) { return 16 * hi + idx; }

__device__ __forceinline__ float leakyf(float x) { return x >= 0.f ? x : ALPHA * x; }

// ---------------------------------------------------------------------------
// Repack a row-major [R x 64] f32 matrix into f16 B-fragments:
// frag index ((kt*4 + nt)*32 + lane), 16 halves each, k = kt*32 + koffA.
// ---------------------------------------------------------------------------
__global__ __launch_bounds__(256) void repack_B_kernel(
    const float* __restrict__ src, h16* __restrict__ dst, int R) {
  int t = blockIdx.x * 256 + threadIdx.x;
  int lane = t & 31, nt = (t >> 5) & 3, kt = t >> 7;
  if (kt * 32 >= R) return;
  int r = lane & 15, hi = lane >> 4;
  int n = nt * 16 + r;
  v16h f;
#pragma unroll
  for (int idx = 0; idx < 16; ++idx) {
    int k = kt * 32 + koffA(hi, idx);
    f[idx] = (h16)src[(size_t)k * DD + n];
  }
  *(v16h*)(dst + (size_t)t * 16) = f;
}

// ---------------------------------------------------------------------------
// Projection GEMM: H[M x 64] = X[M x K] @ W[K x 64], W pre-repacked as frags.
// One wave per 16-row tile; 4 C tiles (full 64 cols); K loop step 32.
// ---------------------------------------------------------------------------
__global__ __launch_bounds__(256) void proj_kernel(
    const float* __restrict__ X, const h16* __restrict__ Wf,
    float* __restrict__ H, int M, int K) {
  int wave = threadIdx.x >> 5, lane = threadIdx.x & 31;
  int m0 = (blockIdx.x * 8 + wave) * 16;
  if (m0 >= M) return;
  int r = lane & 15, hi = lane >> 4;

  // Constant-stride streams: X runs (+0/+16/+64/+80 bytes), W frags (+nt*1024).
  const char* xb = (const char*)(X + (size_t)(m0 + r) * K + 8 * hi);
  const char* wb = (const char*)Wf + (size_t)lane * 32;

  v8f acc[4] = {};
  for (int kk = 0; kk < K; kk += 32, xb += 128, wb += 4096) {
    v4f x0 = *(const v4f*)(xb +  0);
    v4f x1 = *(const v4f*)(xb + 16);
    v4f x2 = *(const v4f*)(xb + 64);
    v4f x3 = *(const v4f*)(xb + 80);
    v16h a;
#pragma unroll
    for (int q = 0; q < 4; ++q) {
      a[q + 0]  = (h16)x0[q];
      a[q + 4]  = (h16)x1[q];
      a[q + 8]  = (h16)x2[q];
      a[q + 12] = (h16)x3[q];
    }
#pragma unroll
    for (int nt = 0; nt < 4; ++nt) {
      v16h b = *(const v16h*)(wb + nt * 1024);
      acc[nt] = __builtin_amdgcn_wmma_f32_16x16x32_f16(
          false, a, false, b, (short)0, acc[nt], false, false);
    }
  }
  // C layout: VGPR v -> row M = v + 8*hi, col N = (lane&15) within 16-tile.
#pragma unroll
  for (int nt = 0; nt < 4; ++nt)
#pragma unroll
    for (int v = 0; v < 8; ++v)
      H[(size_t)(m0 + v + 8 * hi) * DD + nt * 16 + r] = acc[nt][v];
}

// ---------------------------------------------------------------------------
// Per-node scalar scores (4 attention dots) + sigmoid gate.
// ---------------------------------------------------------------------------
__global__ __launch_bounds__(256) void scores_kernel(
    const float* __restrict__ gene_h, const float* __restrict__ cell_h,
    const float* __restrict__ a_gg, const float* __restrict__ a_gc,
    const float* __restrict__ a_cc, const float* __restrict__ a_cg,
    const float* __restrict__ Wgg, const float* __restrict__ bgg,
    const float* __restrict__ Wgc, const float* __restrict__ bgc,
    float* __restrict__ s_gg_src, float* __restrict__ s_gg_dst,
    float* __restrict__ s_gc_src, float* __restrict__ s_cg_dst,
    float* __restrict__ gamma_g,
    float* __restrict__ s_cc_src, float* __restrict__ s_cc_dst,
    float* __restrict__ s_gc_dst, float* __restrict__ s_cg_src,
    float* __restrict__ gamma_c) {
  int t = blockIdx.x * 256 + threadIdx.x;
  if (t < NGn) {
    const float* h = gene_h + (size_t)t * DD;
    float d0 = 0, d1 = 0, d2 = 0, d3 = 0, dg = 0;
#pragma unroll 8
    for (int k = 0; k < DD; ++k) {
      float hv = h[k];
      d0 += hv * a_gg[k];      d1 += hv * a_gg[DD + k];
      d2 += hv * a_gc[k];      d3 += hv * a_cg[DD + k];
      dg += hv * Wgg[k];
    }
    s_gg_src[t] = d0; s_gg_dst[t] = d1; s_gc_src[t] = d2; s_cg_dst[t] = d3;
    gamma_g[t] = 1.f / (1.f + __expf(-(dg + bgg[0])));
  } else if (t < NGn + NCn) {
    int j = t - NGn;
    const float* h = cell_h + (size_t)j * DD;
    float d0 = 0, d1 = 0, d2 = 0, d3 = 0, dg = 0;
#pragma unroll 8
    for (int k = 0; k < DD; ++k) {
      float hv = h[k];
      d0 += hv * a_cc[k];      d1 += hv * a_cc[DD + k];
      d2 += hv * a_gc[DD + k]; d3 += hv * a_cg[k];
      dg += hv * Wgc[k];
    }
    s_cc_src[j] = d0; s_cc_dst[j] = d1; s_gc_dst[j] = d2; s_cg_src[j] = d3;
    gamma_c[j] = 1.f / (1.f + __expf(-(dg + bgc[0])));
  }
}

// ---------------------------------------------------------------------------
// Fused masked-softmax attention: out[i,:] = sum_j w_ij * h_dst[j,:] / sum_j w_ij
// with w_ij = adj_ij ? exp(leaky(s1_i + s2_j)) : 0.
// All 4 relations in one launch (tile ranges) for occupancy.
// ---------------------------------------------------------------------------
struct RelDesc {
  const int* adj;
  const float* s1;
  const float* s2;
  const h16* Bf;     // h_dst pre-repacked as B fragments
  float* out;
  int Nd, lda, trans, tile_start;
};
struct RelPack { RelDesc r[4]; };

__global__ __launch_bounds__(256) void relation_kernel(RelPack P) {
  int wave = threadIdx.x >> 5, lane = threadIdx.x & 31;
  int gtile = blockIdx.x * 8 + wave;
  int rel = 3;
  if      (gtile < P.r[1].tile_start) rel = 0;
  else if (gtile < P.r[2].tile_start) rel = 1;
  else if (gtile < P.r[3].tile_start) rel = 2;
  RelDesc d = P.r[rel];
  int i0 = (gtile - d.tile_start) * 16;
  int r = lane & 15, hi = lane >> 4;

  float s1r = d.s1[i0 + r];

  // Constant-stride streams.
  const char* ab = (const char*)(d.adj + (size_t)(i0 + r) * d.lda + 8 * hi); // +128B/iter
  const char* sb = (const char*)(d.s2 + 8 * hi);                            // +128B/iter
  const char* bb = (const char*)d.Bf + (size_t)lane * 32;                   // +4096B/iter
  const int*  ct = d.adj + (i0 + r);          // transposed path: column base
  const size_t cstride = (size_t)d.lda;       // ints per j-row

  v8f acc[4] = {};
  float psum = 0.f;
  const int niter = d.Nd >> 5;

  for (int it = 0; it < niter; ++it, sb += 128, bb += 4096) {
    v4i a0, a1, a2, a3;
    if (d.trans) {
      // adj^T element (i,j) = adj[j*lda + i]; lanes 0..15 read 64B runs.
      const int* c0 = ct + (size_t)(8 * hi) * cstride;
      const int* c1 = ct + (size_t)(16 + 8 * hi) * cstride;
#pragma unroll
      for (int q = 0; q < 4; ++q) {
        a0[q] = c0[(size_t)q * cstride];
        a1[q] = c0[(size_t)(q + 4) * cstride];
        a2[q] = c1[(size_t)q * cstride];
        a3[q] = c1[(size_t)(q + 4) * cstride];
      }
      ct += cstride * 32;
    } else {
      if (it + 9 < niter)
        __builtin_prefetch(ab + 1024, 0, 1);  // global_prefetch_b8, 8 tiles ahead
      a0 = *(const v4i*)(ab +  0);
      a1 = *(const v4i*)(ab + 16);
      a2 = *(const v4i*)(ab + 64);
      a3 = *(const v4i*)(ab + 80);
      ab += 128;
    }
    v4f s0 = *(const v4f*)(sb +  0);
    v4f s1v = *(const v4f*)(sb + 16);
    v4f s2v = *(const v4f*)(sb + 64);
    v4f s3v = *(const v4f*)(sb + 80);

    // Build attention A-fragment in registers (f16), accumulate row sum.
    v16h a;
#pragma unroll
    for (int q = 0; q < 4; ++q) {
      float e0 = s1r + s0[q];  e0 = e0 >= 0.f ? e0 : ALPHA * e0;
      float e1 = s1r + s1v[q]; e1 = e1 >= 0.f ? e1 : ALPHA * e1;
      float e2 = s1r + s2v[q]; e2 = e2 >= 0.f ? e2 : ALPHA * e2;
      float e3 = s1r + s3v[q]; e3 = e3 >= 0.f ? e3 : ALPHA * e3;
      float w0 = (a0[q] > 0) ? __expf(e0) : 0.f;
      float w1 = (a1[q] > 0) ? __expf(e1) : 0.f;
      float w2 = (a2[q] > 0) ? __expf(e2) : 0.f;
      float w3 = (a3[q] > 0) ? __expf(e3) : 0.f;
      psum += w0 + w1 + w2 + w3;
      a[q + 0]  = (h16)w0;
      a[q + 4]  = (h16)w1;
      a[q + 8]  = (h16)w2;
      a[q + 12] = (h16)w3;
    }
#pragma unroll
    for (int nt = 0; nt < 4; ++nt) {
      v16h b = *(const v16h*)(bb + nt * 1024);
      acc[nt] = __builtin_amdgcn_wmma_f32_16x16x32_f16(
          false, a, false, b, (short)0, acc[nt], false, false);
    }
  }

  // Row-sum reduction: lanes L and L+16 cover disjoint K halves of row L&15.
  float tot = psum + __shfl_xor(psum, 16, 32);
  float inv[8];
#pragma unroll
  for (int v = 0; v < 8; ++v) {
    float rs = __shfl(tot, 8 * hi + v, 32);  // rowsum of M = v + 8*hi
    inv[v] = 1.f / fmaxf(rs, 1e-30f);
  }
#pragma unroll
  for (int nt = 0; nt < 4; ++nt)
#pragma unroll
    for (int v = 0; v < 8; ++v)
      d.out[(size_t)(i0 + v + 8 * hi) * DD + nt * 16 + r] = acc[nt][v] * inv[v];
}

// ---------------------------------------------------------------------------
// Epilogue: gated residual + leaky relu into d_out (gene_out ++ cell_out).
// ---------------------------------------------------------------------------
__global__ __launch_bounds__(256) void combine_kernel(
    const float* __restrict__ gene_h, const float* __restrict__ cell_h,
    const float* __restrict__ gfg, const float* __restrict__ gfc,
    const float* __restrict__ cfc, const float* __restrict__ cfg,
    const float* __restrict__ gamma_g, const float* __restrict__ gamma_c,
    float* __restrict__ out) {
  int t = blockIdx.x * 256 + threadIdx.x;
  const int GN = NGn * DD;
  if (t < GN) {
    int i = t >> 6;
    out[t] = leakyf(gene_h[t] + gfg[t] + gamma_g[i] * gfc[t]);
  } else if (t < GN + NCn * DD) {
    int u = t - GN;
    int i = u >> 6;
    out[t] = leakyf(cell_h[u] + cfc[u] + gamma_c[i] * cfg[u]);
  }
}

// ---------------------------------------------------------------------------
extern "C" void kernel_launch(void* const* d_in, const int* in_sizes, int n_in,
                              void* d_out, int out_size, void* d_ws, size_t ws_size,
                              hipStream_t stream) {
  const float* gene_x  = (const float*)d_in[0];
  const float* cell_x  = (const float*)d_in[1];
  const float* W_g     = (const float*)d_in[2];
  const float* W_c     = (const float*)d_in[3];
  const float* a_gg    = (const float*)d_in[4];
  const float* a_gc    = (const float*)d_in[5];
  const float* a_cc    = (const float*)d_in[6];
  const float* a_cg    = (const float*)d_in[7];
  const float* Wgate_g = (const float*)d_in[8];
  const float* bgate_g = (const float*)d_in[9];
  const float* Wgate_c = (const float*)d_in[10];
  const float* bgate_c = (const float*)d_in[11];
  const int* gene_adj      = (const int*)d_in[12];
  const int* cell_adj      = (const int*)d_in[13];
  const int* gene_cell_adj = (const int*)d_in[14];
  float* out = (float*)d_out;

  char* p = (char*)d_ws;
  auto alloc = [&](size_t bytes) -> char* {
    char* q = p;
    p += (bytes + 255) & ~(size_t)255;
    return q;
  };
  float* gene_h = (float*)alloc((size_t)NGn * DD * 4);
  float* cell_h = (float*)alloc((size_t)NCn * DD * 4);
  h16* geneF = (h16*)alloc((size_t)NGn * DD * 2);
  h16* cellF = (h16*)alloc((size_t)NCn * DD * 2);
  h16* WgF   = (h16*)alloc((size_t)(FGk / 32) * 128 * 16 * 2);
  h16* WcF   = (h16*)alloc((size_t)(FCk / 32) * 128 * 16 * 2);
  float* s_gg_src = (float*)alloc(NGn * 4);
  float* s_gg_dst = (float*)alloc(NGn * 4);
  float* s_gc_src = (float*)alloc(NGn * 4);
  float* s_cg_dst = (float*)alloc(NGn * 4);
  float* gamma_g  = (float*)alloc(NGn * 4);
  float* s_cc_src = (float*)alloc(NCn * 4);
  float* s_cc_dst = (float*)alloc(NCn * 4);
  float* s_gc_dst = (float*)alloc(NCn * 4);
  float* s_cg_src = (float*)alloc(NCn * 4);
  float* gamma_c  = (float*)alloc(NCn * 4);
  float* gfg = (float*)alloc((size_t)NGn * DD * 4);
  float* gfc = (float*)alloc((size_t)NGn * DD * 4);
  float* cfc = (float*)alloc((size_t)NCn * DD * 4);
  float* cfg = (float*)alloc((size_t)NCn * DD * 4);
  (void)ws_size; (void)in_sizes; (void)n_in; (void)out_size;

  // 1) Repack weight matrices into B fragments.
  repack_B_kernel<<<(FGk / 32) * 128 / 256, 256, 0, stream>>>(W_g, WgF, FGk);
  repack_B_kernel<<<(FCk / 32) * 128 / 256, 256, 0, stream>>>(W_c, WcF, FCk);

  // 2) Projection GEMMs (f16 WMMA, f32 accumulation).
  proj_kernel<<<NGn / 128, 256, 0, stream>>>(gene_x, WgF, gene_h, NGn, FGk);
  proj_kernel<<<NCn / 128, 256, 0, stream>>>(cell_x, WcF, cell_h, NCn, FCk);

  // 3) Repack h matrices as B fragments for the attention GEMMs.
  repack_B_kernel<<<(NGn / 32) * 128 / 256, 256, 0, stream>>>(gene_h, geneF, NGn);
  repack_B_kernel<<<(NCn / 32) * 128 / 256, 256, 0, stream>>>(cell_h, cellF, NCn);

  // 4) Score vectors + gates.
  scores_kernel<<<(NGn + NCn) / 256, 256, 0, stream>>>(
      gene_h, cell_h, a_gg, a_gc, a_cc, a_cg, Wgate_g, bgate_g, Wgate_c, bgate_c,
      s_gg_src, s_gg_dst, s_gc_src, s_cg_dst, gamma_g,
      s_cc_src, s_cc_dst, s_gc_dst, s_cg_src, gamma_c);

  // 5) All four attention relations fused in one launch.
  RelPack pk;
  pk.r[0] = {gene_adj,      s_gg_src, s_gg_dst, geneF, gfg, NGn, NGn, 0, 0};
  pk.r[1] = {gene_cell_adj, s_gc_src, s_gc_dst, cellF, gfc, NCn, NCn, 0, NGn / 16};
  pk.r[2] = {cell_adj,      s_cc_src, s_cc_dst, cellF, cfc, NCn, NCn, 0, NGn / 16 + NGn / 16};
  pk.r[3] = {gene_cell_adj, s_cg_src, s_cg_dst, geneF, cfg, NGn, NCn, 1,
             NGn / 16 + NGn / 16 + NCn / 16};
  int total_tiles = NGn / 16 + NGn / 16 + NCn / 16 + NCn / 16;  // 1536
  relation_kernel<<<total_tiles / 8, 256, 0, stream>>>(pk);

  // 6) Gated residual epilogue.
  combine_kernel<<<(NGn + NCn) * DD / 256, 256, 0, stream>>>(
      gene_h, cell_h, gfg, gfc, cfc, cfg, gamma_g, gamma_c, out);
}
